// Attention_50921132262075
// MI455X (gfx1250) — matrix operand order repaired and checked by reading, not compiled, not run on hardware
//
#include <hip/hip_runtime.h>
#include <hip/hip_bf16.h>
#include <math.h>

// ---------------------------------------------------------------------------
// Fused flash-attention for B=4, C=256, C8=32, N=4096 on gfx1250 (wave32).
//   kernel 1/2: 1x1-conv projections -> f16 q/k/v in WMMA-friendly layouts
//   kernel 3  : fused  S = q^T k  -> online softmax -> O += P v^T
// v3: K/V tiles double-buffered in LDS via global_load_async_to_lds_b128
//     (ASYNCcnt) -> ~5x cut in L2 fragment traffic, copy/compute overlap.
// ---------------------------------------------------------------------------

typedef __attribute__((ext_vector_type(16))) _Float16 v16h;
typedef __attribute__((ext_vector_type(8)))  float    v8f;

#define N_TOK 4096
#define C_DIM 256
#define C8    32
#define BM    64      // query rows per workgroup (4 i-tiles x 16)
#define BN    64      // key/value columns per main-loop step

// LDS row strides (dwords), chosen so the two 16-lane halves of a fragment
// load hit disjoint bank runs (stride%8==4 -> 8 rows shift banks by 32) while
// staying 16B-aligned for async B128 stores.
#define VSTR  260     // V tile rows: 32 j-pairs x 256 c-dwords (+4 pad)
#define KSTR  68      // K tile rows: 16 c-pairs x 64 j-dwords  (+4 pad)
#define PSTR  66      // P rows in halfs (33 dwords, odd -> conflict-free)

union Frag16 { v16h h; unsigned int u[8]; };

__device__ __forceinline__ void async_ld128(unsigned lds_off, const void* g) {
    // CDNA5 async DMA: LDS[lds_off] = MEM[g] (16B), tracked by ASYNCcnt.
    asm volatile("global_load_async_to_lds_b128 %0, %1, off"
                 :: "v"(lds_off), "v"(g) : "memory");
}

// ---------------------------------------------------------------------------
// Projection: q = Wq x + bq (sel 0), k = Wk y + bk (sel 1).
// qT layout: [b][i][c8]           (A-fragment rows, c-pairs contiguous)
// kP layout: [b][c8/2][j][2]      (B-fragment K-pairs packed per dword)
// ---------------------------------------------------------------------------
__global__ __launch_bounds__(256) void proj_qk(
    const float* __restrict__ x, const float* __restrict__ y,
    const float* __restrict__ Wq, const float* __restrict__ bq,
    const float* __restrict__ Wk, const float* __restrict__ bk,
    _Float16* __restrict__ qT, _Float16* __restrict__ kP)
{
    const int i   = blockIdx.x * 256 + threadIdx.x;   // token (coalesced)
    const int o   = blockIdx.y & 31;                  // out channel (0..31)
    const int sel = blockIdx.y >> 5;                  // 0 = q, 1 = k
    const int b   = blockIdx.z;

    const float* in = (sel == 0 ? x : y) + (size_t)b * C_DIM * N_TOK + i;
    const float* W  = (sel == 0 ? Wq : Wk) + (size_t)o * C_DIM;
    float acc = (sel == 0 ? bq[o] : bk[o]);
    #pragma unroll 8
    for (int c = 0; c < C_DIM; ++c) acc += W[c] * in[(size_t)c * N_TOK];

    const _Float16 hv = (_Float16)acc;
    if (sel == 0)
        qT[((size_t)b * N_TOK + i) * C8 + o] = hv;
    else
        kP[(((size_t)b * (C8 / 2) + (o >> 1)) * N_TOK + i) * 2 + (o & 1)] = hv;
}

// ---------------------------------------------------------------------------
// Projection: v = Wv y + bv.
// vP layout: [b][j/2][c][2]  -> B-fragment K(=j)-pairs packed per dword,
//                              c contiguous so staging rows are coalesced.
// ---------------------------------------------------------------------------
__global__ __launch_bounds__(256) void proj_v(
    const float* __restrict__ y, const float* __restrict__ Wv,
    const float* __restrict__ bv, _Float16* __restrict__ vP)
{
    const int i = blockIdx.x * 256 + threadIdx.x;
    const int o = blockIdx.y;
    const int b = blockIdx.z;

    const float* in = y + (size_t)b * C_DIM * N_TOK + i;
    const float* W  = Wv + (size_t)o * C_DIM;
    float acc = bv[o];
    #pragma unroll 8
    for (int c = 0; c < C_DIM; ++c) acc += W[c] * in[(size_t)c * N_TOK];

    vP[(((size_t)b * (N_TOK / 2) + (i >> 1)) * C_DIM + o) * 2 + (i & 1)] =
        (_Float16)acc;
}

// ---------------------------------------------------------------------------
// Fused attention. Grid (N/BM, B), 256 threads = 8 wave32 waves.
// Wave w: i-tile = w>>1 (16 rows), channel half = w&1 (128 of 256 channels).
// K/V j-block tiles are staged cooperatively into double-buffered LDS with
// async copies; fragments are then gathered from LDS (WGP-local bandwidth).
// ---------------------------------------------------------------------------
__global__ __launch_bounds__(256) void attn_flash(
    const float* __restrict__ x, const float* __restrict__ gamma,
    const _Float16* __restrict__ qT, const _Float16* __restrict__ kP,
    const _Float16* __restrict__ vP, float* __restrict__ out)
{
    __shared__ __align__(16) unsigned  vds[2][32 * VSTR];   // 66,560 B
    __shared__ __align__(16) unsigned  kds[2][16 * KSTR];   //  8,704 B
    __shared__ __align__(16) _Float16  Pbuf[8 * 16 * PSTR]; // 16,896 B

    const int b      = blockIdx.y;
    const int rowblk = blockIdx.x;
    const int wave   = threadIdx.x >> 5;
    const int lane   = threadIdx.x & 31;
    const int it     = wave >> 1;
    const int ch     = wave & 1;          // channel half: c base = ch*128
    const int i0     = rowblk * BM + it * 16;
    const int row    = lane & 15;
    const int hlf    = lane >> 4;

    const unsigned* q32 = (const unsigned*)qT +
                          ((size_t)b * N_TOK + i0 + row) * (C8 / 2);
    const unsigned* kg  = (const unsigned*)kP + (size_t)b * (C8 / 2) * N_TOK;
    const unsigned* vg  = (const unsigned*)vP + (size_t)b * (N_TOK / 2) * C_DIM;

    // Cooperative async staging of one j-block: V = 32x256 dwords (8 B128
    // chunks/thread), K = 16x64 dwords (1 chunk/thread) -> 9 async ops/thread.
    auto stage = [&](int jb, int bufn) {
        #pragma unroll
        for (int t = 0; t < 8; ++t) {
            const int idx = t * 256 + (int)threadIdx.x;   // 0..2047
            const int jj  = idx >> 6;                     // j-pair row 0..31
            const int c4  = (idx & 63) << 2;              // dword col, x4
            async_ld128((unsigned)(unsigned long long)&vds[bufn][jj * VSTR + c4],
                        vg + (size_t)((jb >> 1) + jj) * C_DIM + c4);
        }
        const int cc = (int)threadIdx.x >> 4;             // c-pair row 0..15
        const int j4 = ((int)threadIdx.x & 15) << 2;      // dword col, x4
        async_ld128((unsigned)(unsigned long long)&kds[bufn][cc * KSTR + j4],
                    kg + (size_t)cc * N_TOK + jb + j4);
    };

    // Q A-fragment (16x32 f16), ISA layout: lanes<16 K=0..7,16..23; lanes>=16
    // K=8..15,24..31, two K per VGPR. Lives in registers for the whole kernel.
    Frag16 qa;
    #pragma unroll
    for (int v = 0; v < 8; ++v)
        qa.u[v] = q32[hlf * 4 + (v < 4 ? v : v + 4)];

    float m[8], lsum[8];
    v8f acc[8];
    #pragma unroll
    for (int r = 0; r < 8; ++r) { m[r] = -1.0e30f; lsum[r] = 0.0f; }
    #pragma unroll
    for (int ct = 0; ct < 8; ++ct) { v8f z = {}; acc[ct] = z; }

    stage(0, 0);  // prologue fill of buffer 0

    for (int jb = 0; jb < N_TOK; jb += BN) {
        const int buf = (jb >> 6) & 1;
        const bool hasNext = (jb + BN) < N_TOK;
        if (hasNext) {
            stage(jb + BN, buf ^ 1);   // overlap next block's DMA with compute
            // older 9 (current buffer) done; newer 9 may remain in flight
            asm volatile("s_wait_asynccnt 0x9" ::: "memory");
        } else {
            asm volatile("s_wait_asynccnt 0x0" ::: "memory");
        }
        __syncthreads();               // all waves' staged data visible

        const unsigned* kt = &kds[buf][0];
        const unsigned* vt = &vds[buf][0];

        // ---- S = q^T k for 4 j-subtiles; K=32 => one WMMA per tile ----
        v8f S[4];
        #pragma unroll
        for (int t = 0; t < 4; ++t) {
            Frag16 kb;  // B-frag 32x16: lanes<16 K=0..7,16..23; lanes>=16 K=8..15,24..31
            #pragma unroll
            for (int v = 0; v < 8; ++v)
                kb.u[v] = kt[(hlf * 4 + (v < 4 ? v : v + 4)) * KSTR + t * 16 + row];
            v8f z = {};
            S[t] = __builtin_amdgcn_wmma_f32_16x16x32_f16(
                false, qa.h, false, kb.h, (short)0, z, false, false);
        }

        // ---- online softmax (rows striped: row = r + 8*hlf, col = lane&15) --
        #pragma unroll
        for (int r = 0; r < 8; ++r) {
            float bm = fmaxf(fmaxf(S[0][r], S[1][r]), fmaxf(S[2][r], S[3][r]));
            bm = fmaxf(bm, __shfl_xor(bm, 1, 32));
            bm = fmaxf(bm, __shfl_xor(bm, 2, 32));
            bm = fmaxf(bm, __shfl_xor(bm, 4, 32));
            bm = fmaxf(bm, __shfl_xor(bm, 8, 32));   // reduce within 16-lane half
            const float mn = fmaxf(m[r], bm);
            const float sc = __expf(m[r] - mn);
            m[r] = mn;
            lsum[r] *= sc;
            #pragma unroll
            for (int ct = 0; ct < 8; ++ct) acc[ct][r] *= sc;  // rescale O
            float ps = 0.0f;
            #pragma unroll
            for (int t = 0; t < 4; ++t) {
                const float p = __expf(S[t][r] - mn);
                ps += p;
                Pbuf[(wave * 16 + r + 8 * hlf) * PSTR + t * 16 + row] =
                    (_Float16)p;
            }
            ps += __shfl_xor(ps, 1, 32);
            ps += __shfl_xor(ps, 2, 32);
            ps += __shfl_xor(ps, 4, 32);
            ps += __shfl_xor(ps, 8, 32);
            lsum[r] += ps;
        }
        __syncthreads();  // uniform; orders P write -> A-frag read

        // ---- O += P v^T : 2 K-fragments x 8 channel tiles ----
        #pragma unroll
        for (int kf = 0; kf < 2; ++kf) {
            Frag16 pa;
            const unsigned* p32 =
                (const unsigned*)Pbuf + (wave * 16 + row) * (PSTR / 2);
            #pragma unroll
            for (int v = 0; v < 8; ++v)
                pa.u[v] = p32[kf * 16 + hlf * 4 + (v < 4 ? v : v + 4)];
            #pragma unroll
            for (int ct = 0; ct < 8; ++ct) {
                Frag16 vb;
                const int c = ch * 128 + ct * 16 + row;
                #pragma unroll
                for (int v = 0; v < 8; ++v)
                    vb.u[v] = vt[(kf * 16 + hlf * 4 + (v < 4 ? v : v + 4)) * VSTR + c];
                acc[ct] = __builtin_amdgcn_wmma_f32_16x16x32_f16(
                    false, pa.h, false, vb.h, (short)0, acc[ct], false, false);
            }
        }
        __syncthreads();  // protect Pbuf + LDS buffers before next staging
    }

    // ---- epilogue: out = gamma * (O / l) + x ----
    const float g = gamma[0];
    #pragma unroll
    for (int r = 0; r < 8; ++r) {
        const float inv = 1.0f / lsum[r];
        const int i = i0 + r + 8 * hlf;
        #pragma unroll
        for (int ct = 0; ct < 8; ++ct) {
            const int c = ch * 128 + ct * 16 + row;
            const size_t idx = ((size_t)b * C_DIM + c) * N_TOK + i;
            out[idx] = g * acc[ct][r] * inv + x[idx];
        }
    }
}

// ---------------------------------------------------------------------------
extern "C" void kernel_launch(void* const* d_in, const int* in_sizes, int n_in,
                              void* d_out, int out_size, void* d_ws, size_t ws_size,
                              hipStream_t stream) {
    const float* x     = (const float*)d_in[0];
    const float* y     = (const float*)d_in[1];
    const float* Wq    = (const float*)d_in[2];
    const float* bq    = (const float*)d_in[3];
    const float* Wk    = (const float*)d_in[4];
    const float* bk    = (const float*)d_in[5];
    const float* Wv    = (const float*)d_in[6];
    const float* bv    = (const float*)d_in[7];
    const float* gamma = (const float*)d_in[8];
    float* out = (float*)d_out;

    // workspace: qT 1MB | kP 1MB | vP 8MB  (f16)
    char* ws = (char*)d_ws;
    _Float16* qT = (_Float16*)(ws);
    _Float16* kP = (_Float16*)(ws + (1u << 20));
    _Float16* vP = (_Float16*)(ws + (2u << 20));

    dim3 blk(256);
    proj_qk<<<dim3(N_TOK / 256, 64, 4), blk, 0, stream>>>(
        x, y, Wq, bq, Wk, bk, qT, kP);
    proj_v<<<dim3(N_TOK / 256, C_DIM, 4), blk, 0, stream>>>(y, Wv, bv, vP);
    attn_flash<<<dim3(N_TOK / BM, 4), blk, 0, stream>>>(
        x, gamma, qT, kP, vP, out);
}
